// MultiHeadAttention_74517682586039
// MI455X (gfx1250) — matrix operand order repaired and checked by reading, not compiled, not run on hardware
//
#include <hip/hip_runtime.h>
#include <hip/hip_bf16.h>

#define D_MODEL 1024
#define S_LEN   2048
#define NHEAD   16
#define DHEAD   64
#define BATCH   4

typedef __bf16 bf16_t;
typedef bf16_t v16bf __attribute__((ext_vector_type(16)));
typedef bf16_t v2bf  __attribute__((ext_vector_type(2)));
typedef float  v8f   __attribute__((ext_vector_type(8)));
typedef unsigned int v4u __attribute__((ext_vector_type(4)));
typedef float  v4f   __attribute__((ext_vector_type(4)));
typedef int    v4i   __attribute__((ext_vector_type(4)));

typedef __attribute__((address_space(1))) v4i* gptr_v4i;   // global int4*
typedef __attribute__((address_space(3))) v4i* lptr_v4i;   // LDS int4*

#if defined(__has_builtin)
#if __has_builtin(__builtin_amdgcn_global_load_async_to_lds_b128)
#define HAVE_ASYNC_LDS 1
#endif
#endif
#ifndef HAVE_ASYNC_LDS
#define HAVE_ASYNC_LDS 0
#endif

union AFrag {
    v16bf v;
    v4u   q[2];
};

// hardware bf16 convert (v_cvt_pk_bf16_f32), RNE
__device__ __forceinline__ unsigned int pk_bf16(float a, float b) {
    union { v2bf h; unsigned int u; } cv;
    cv.h = v2bf{(bf16_t)a, (bf16_t)b};
    return cv.u;
}
__device__ __forceinline__ unsigned short f32_to_bf16(float f) {
    union { bf16_t h; unsigned short u; } cv;
    cv.h = (bf16_t)f;
    return cv.u;
}

__device__ __forceinline__ v8f wmma_bf16(v16bf a, v16bf b, v8f c) {
    return __builtin_amdgcn_wmma_f32_16x16x32_bf16(false, a, false, b, (short)0, c,
                                                   false, false);
}
// identical opcode, same A registers as the immediately preceding WMMA -> set
// the VOP3P reuse-A hint (operand cache in the matrix pipe).
__device__ __forceinline__ v8f wmma_bf16_ra(v16bf a, v16bf b, v8f c) {
    return __builtin_amdgcn_wmma_f32_16x16x32_bf16(false, a, false, b, (short)0, c,
                                                   true, false);
}

// 16-byte global->LDS copy: async-to-LDS (ASYNCcnt path) when available
__device__ __forceinline__ void copy16_g2l(const unsigned short* g, unsigned short* l) {
#if HAVE_ASYNC_LDS
    __builtin_amdgcn_global_load_async_to_lds_b128(
        (gptr_v4i)(unsigned long long)g,
        (lptr_v4i)(unsigned int)(unsigned long long)l, 0, 0);
#else
    *(v4u*)l = *(const v4u*)g;
#endif
}
__device__ __forceinline__ void copy_fence() {
#if HAVE_ASYNC_LDS
    asm volatile("s_wait_asynccnt 0" ::: "memory");
#endif
}

// ---------------------------------------------------------------------------
// Projection GEMM: Y = X(f32)[8192,1024] @ W^T(f32)[1024,1024] + b -> bf16.
// VT == 0 : Out[B,H,S,DHEAD]   (Q, K)
// VT == 1 : Out[B,H,DHEAD,S]   (V; makes attention V tiles contiguous)
// Block: 256 threads = 8 waves. BM=128, BN=128, BK=32; 8 WMMA / wave / k-step.
// ---------------------------------------------------------------------------
template <int VT>
__global__ void __launch_bounds__(256) proj_qkv_kernel(
    const float* __restrict__ X, const float* __restrict__ W,
    const float* __restrict__ bias, unsigned short* __restrict__ Out) {
    __shared__ __attribute__((aligned(16))) unsigned short As[128 * 32];
    __shared__ __attribute__((aligned(16))) unsigned short Bs[128 * 32];

    const int tid  = threadIdx.x;
    const int lane = tid & 31;
    const int wave = tid >> 5;
    const int m0   = blockIdx.x * 128;
    const int n0   = blockIdx.y * 128;

    const int mrow  = lane & 15;
    const int c0    = (lane < 16) ? 0 : 8;
    const int kbase = (lane < 16) ? 0 : 16;

    v8f acc[8];
    const v8f vzero = {0.f, 0.f, 0.f, 0.f, 0.f, 0.f, 0.f, 0.f};
#pragma unroll
    for (int t = 0; t < 8; ++t) acc[t] = vzero;

    const int row = tid >> 1;
    const int ks  = (tid & 1) * 16;

    for (int k0 = 0; k0 < D_MODEL; k0 += 32) {
        // stage X tile 128x32 (f32 -> bf16 via v_cvt_pk_bf16_f32)
        {
            const float* src = X + (size_t)(m0 + row) * D_MODEL + k0 + ks;
            v4f f0 = *(const v4f*)(src + 0);
            v4f f1 = *(const v4f*)(src + 4);
            v4f f2 = *(const v4f*)(src + 8);
            v4f f3 = *(const v4f*)(src + 12);
            v4u d0 = {pk_bf16(f0.x, f0.y), pk_bf16(f0.z, f0.w),
                      pk_bf16(f1.x, f1.y), pk_bf16(f1.z, f1.w)};
            v4u d1 = {pk_bf16(f2.x, f2.y), pk_bf16(f2.z, f2.w),
                      pk_bf16(f3.x, f3.y), pk_bf16(f3.z, f3.w)};
            *(v4u*)(As + row * 32 + ks)     = d0;
            *(v4u*)(As + row * 32 + ks + 8) = d1;
        }
        // stage W tile 128x32
        {
            const float* src = W + (size_t)(n0 + row) * D_MODEL + k0 + ks;
            v4f f0 = *(const v4f*)(src + 0);
            v4f f1 = *(const v4f*)(src + 4);
            v4f f2 = *(const v4f*)(src + 8);
            v4f f3 = *(const v4f*)(src + 12);
            v4u d0 = {pk_bf16(f0.x, f0.y), pk_bf16(f0.z, f0.w),
                      pk_bf16(f1.x, f1.y), pk_bf16(f1.z, f1.w)};
            v4u d1 = {pk_bf16(f2.x, f2.y), pk_bf16(f2.z, f2.w),
                      pk_bf16(f3.x, f3.y), pk_bf16(f3.z, f3.w)};
            *(v4u*)(Bs + row * 32 + ks)     = d0;
            *(v4u*)(Bs + row * 32 + ks + 8) = d1;
        }
        __syncthreads();

        // preload A fragment + all 8 B fragments, then a clean WMMA burst
        AFrag fa;
        fa.q[0] = *(const v4u*)(As + (wave * 16 + mrow) * 32 + c0);
        fa.q[1] = *(const v4u*)(As + (wave * 16 + mrow) * 32 + 16 + c0);
        AFrag fb[8];
#pragma unroll
        for (int t = 0; t < 8; ++t) {
            const v4u* p = (const v4u*)(Bs + (t * 16 + mrow) * 32 + c0);
            fb[t].q[0] = p[0];
            fb[t].q[1] = p[2];
        }
        acc[0] = wmma_bf16(fa.v, fb[0].v, acc[0]);
#pragma unroll
        for (int t = 1; t < 8; ++t)
            acc[t] = wmma_bf16_ra(fa.v, fb[t].v, acc[t]);   // same A as previous
        __syncthreads();
    }

    // epilogue: bias + bf16 store (head-split layouts)
#pragma unroll
    for (int t = 0; t < 8; ++t) {
        const int ncol = n0 + t * 16 + mrow;
        const float bv = bias[ncol];
        const int h = ncol >> 6;
        const int d = ncol & (DHEAD - 1);
#pragma unroll
        for (int r = 0; r < 8; ++r) {
            const int m = m0 + wave * 16 + ((lane < 16) ? r : 8 + r);
            const int b = m >> 11;
            const int s = m & (S_LEN - 1);
            const size_t dst =
                VT ? (((size_t)b * NHEAD + h) * DHEAD + d) * S_LEN + s
                   : (((size_t)b * NHEAD + h) * S_LEN + s) * DHEAD + d;
            Out[dst] = f32_to_bf16(acc[t][r] + bv);
        }
    }
}

// ---------------------------------------------------------------------------
// Flash attention: 128 threads = 4 waves, 16 q-rows/wave, kv tiled by 32.
// Qh,Kh: [B,H,S,DHEAD] bf16; Vh: [B,H,DHEAD,S] bf16 (pre-transposed).
// K/V tiles staged with GLOBAL_LOAD_ASYNC_TO_LDS_B128.
// ---------------------------------------------------------------------------
__global__ void __launch_bounds__(128) attn_kernel(
    const unsigned short* __restrict__ Qh, const unsigned short* __restrict__ Kh,
    const unsigned short* __restrict__ Vh, unsigned short* __restrict__ Out) {
    __shared__ __attribute__((aligned(16))) unsigned short Kt[32 * 64];    // [kv][d]
    __shared__ __attribute__((aligned(16))) unsigned short Vt[64 * 32];    // [d][kv]
    __shared__ __attribute__((aligned(16))) unsigned short Pw[4][16 * 32]; // per-wave P

    const int tid  = threadIdx.x;
    const int lane = tid & 31;
    const int wave = tid >> 5;
    const int b = blockIdx.z, h = blockIdx.y;
    const int q0 = blockIdx.x * 64 + wave * 16;
    const size_t head_base = ((size_t)b * NHEAD + h) * (size_t)S_LEN * DHEAD;

    const int mrow  = lane & 15;
    const int c0    = (lane < 16) ? 0 : 8;
    const int kbase = (lane < 16) ? 0 : 16;

    // Q fragments, K-dim 0..63, resident for whole loop
    AFrag qf[2];
    {
        const unsigned short* qrow = Qh + head_base + (size_t)(q0 + mrow) * DHEAD;
#pragma unroll
        for (int jc = 0; jc < 2; ++jc) {
            qf[jc].q[0] = *(const v4u*)(qrow + jc * 32 + c0);
            qf[jc].q[1] = *(const v4u*)(qrow + jc * 32 + 16 + c0);
        }
    }

    const v8f vzero = {0.f, 0.f, 0.f, 0.f, 0.f, 0.f, 0.f, 0.f};
    v8f o[4];
#pragma unroll
    for (int t = 0; t < 4; ++t) o[t] = vzero;
    float mrun[8], lrun[8];
#pragma unroll
    for (int r = 0; r < 8; ++r) { mrun[r] = -3.0e38f; lrun[r] = 0.f; }

    unsigned short* myP = Pw[wave];

    // per-thread copy assignments: two 16B chunks each for K and V tiles
    const int krow0 = tid >> 2;              // K tile rows 0..31
    const int kseg0 = (tid & 3) * 8;
    const int krow1 = (tid + 128) >> 2;
    const int kseg1 = ((tid + 128) & 3) * 8;
    const int vrow0 = tid >> 1;              // V tile rows 0..63 (2 chunks/row)
    const int vseg0 = (tid & 1) * 16;

    for (int kv0 = 0; kv0 < S_LEN; kv0 += 32) {
        // ---- stage K tile [32][64] and V tile [64][32] (both plain row copies)
        copy16_g2l(Kh + head_base + (size_t)(kv0 + krow0) * DHEAD + kseg0,
                   Kt + krow0 * 64 + kseg0);
        copy16_g2l(Kh + head_base + (size_t)(kv0 + krow1) * DHEAD + kseg1,
                   Kt + krow1 * 64 + kseg1);
        copy16_g2l(Vh + head_base + (size_t)vrow0 * S_LEN + kv0 + vseg0,
                   Vt + vrow0 * 32 + vseg0);
        copy16_g2l(Vh + head_base + (size_t)(vrow0 + 32) * S_LEN + kv0 + vseg0,
                   Vt + (vrow0 + 32) * 32 + vseg0);
        if (kv0 + 32 < S_LEN) {
            __builtin_prefetch(Kh + head_base + (size_t)(kv0 + 32 + krow0) * DHEAD, 0, 1);
            __builtin_prefetch(Vh + head_base + (size_t)vrow0 * S_LEN + kv0 + 32, 0, 1);
        }
        copy_fence();
        __syncthreads();

        // ---- preload all 4 K B-fragments, then the score WMMA burst
        AFrag fk[2][2];   // [kv sub-tile][K chunk]
#pragma unroll
        for (int t = 0; t < 2; ++t)
#pragma unroll
            for (int jc = 0; jc < 2; ++jc) {
                const v4u* p = (const v4u*)(Kt + (t * 16 + mrow) * 64 + jc * 32 + c0);
                fk[t][jc].q[0] = p[0];
                fk[t][jc].q[1] = p[2];
            }
        v8f st[2] = {vzero, vzero};
#pragma unroll
        for (int jc = 0; jc < 2; ++jc) {     // jc outer: consecutive WMMAs share A
            st[0] = wmma_bf16(qf[jc].v, fk[0][jc].v, st[0]);
            st[1] = wmma_bf16_ra(qf[jc].v, fk[1][jc].v, st[1]);
        }

        // ---- online softmax (row lives across 16 lanes of each half-wave)
        const float scale = 0.125f;
#pragma unroll
        for (int r = 0; r < 8; ++r) {
            float s0 = st[0][r] * scale;
            float s1 = st[1][r] * scale;
            float vm = fmaxf(s0, s1);
#pragma unroll
            for (int off = 1; off < 16; off <<= 1)
                vm = fmaxf(vm, __shfl_xor(vm, off, 32));
            const float mnew = fmaxf(mrun[r], vm);
            const float corr = __expf(mrun[r] - mnew);
            const float p0 = __expf(s0 - mnew);
            const float p1 = __expf(s1 - mnew);
            float rs = p0 + p1;
#pragma unroll
            for (int off = 1; off < 16; off <<= 1)
                rs += __shfl_xor(rs, off, 32);
            lrun[r] = lrun[r] * corr + rs;
            mrun[r] = mnew;
#pragma unroll
            for (int t2 = 0; t2 < 4; ++t2) o[t2][r] *= corr;
            const int M = (lane < 16) ? r : 8 + r;
            myP[M * 32 + mrow]      = f32_to_bf16(p0);
            myP[M * 32 + 16 + mrow] = f32_to_bf16(p1);
        }
        // same-wave LDS RAW fence for C-layout -> A-layout re-swizzle
        asm volatile("s_wait_dscnt 0" ::: "memory");

        AFrag pf;
        pf.q[0] = *(const v4u*)(myP + mrow * 32 + c0);
        pf.q[1] = *(const v4u*)(myP + mrow * 32 + 16 + c0);

        // ---- O += P[16x32] @ V[32x64]: preload 4 V frags, shared-A burst
        AFrag fv[4];
#pragma unroll
        for (int dt = 0; dt < 4; ++dt) {
            const v4u* p = (const v4u*)(Vt + (dt * 16 + mrow) * 32 + c0);
            fv[dt].q[0] = p[0];
            fv[dt].q[1] = p[2];
        }
        o[0] = wmma_bf16(pf.v, fv[0].v, o[0]);
#pragma unroll
        for (int dt = 1; dt < 4; ++dt)
            o[dt] = wmma_bf16_ra(pf.v, fv[dt].v, o[dt]);
        __syncthreads();
    }

    // ---- normalize + store bf16 [B, S, D_MODEL]
#pragma unroll
    for (int dt = 0; dt < 4; ++dt) {
#pragma unroll
        for (int r = 0; r < 8; ++r) {
            const int M = (lane < 16) ? r : 8 + r;
            const int srow = q0 + M;
            const int d = dt * 16 + mrow;
            const size_t dst = ((size_t)b * S_LEN + srow) * D_MODEL + h * DHEAD + d;
            Out[dst] = f32_to_bf16(o[dt][r] / lrun[r]);
        }
    }
}

// ---------------------------------------------------------------------------
// Output projection: attn(bf16)[8192,1024] @ w_o^T(f32) + b_o -> f32 d_out
// ---------------------------------------------------------------------------
__global__ void __launch_bounds__(256) proj_out_kernel(
    const unsigned short* __restrict__ Xb, const float* __restrict__ W,
    const float* __restrict__ bias, float* __restrict__ Out) {
    __shared__ __attribute__((aligned(16))) unsigned short As[128 * 32];
    __shared__ __attribute__((aligned(16))) unsigned short Bs[128 * 32];

    const int tid  = threadIdx.x;
    const int lane = tid & 31;
    const int wave = tid >> 5;
    const int m0   = blockIdx.x * 128;
    const int n0   = blockIdx.y * 128;

    const int mrow  = lane & 15;
    const int c0    = (lane < 16) ? 0 : 8;
    const int kbase = (lane < 16) ? 0 : 16;

    v8f acc[8];
    const v8f vzero = {0.f, 0.f, 0.f, 0.f, 0.f, 0.f, 0.f, 0.f};
#pragma unroll
    for (int t = 0; t < 8; ++t) acc[t] = vzero;

    const int row = tid >> 1;
    const int ks  = (tid & 1) * 16;

    for (int k0 = 0; k0 < D_MODEL; k0 += 32) {
        {
            // bf16 input: straight async copy into LDS
            const unsigned short* src = Xb + (size_t)(m0 + row) * D_MODEL + k0 + ks;
            copy16_g2l(src, As + row * 32 + ks);
            copy16_g2l(src + 8, As + row * 32 + ks + 8);
        }
        {
            const float* src = W + (size_t)(n0 + row) * D_MODEL + k0 + ks;
            v4f f0 = *(const v4f*)(src + 0);
            v4f f1 = *(const v4f*)(src + 4);
            v4f f2 = *(const v4f*)(src + 8);
            v4f f3 = *(const v4f*)(src + 12);
            v4u d0 = {pk_bf16(f0.x, f0.y), pk_bf16(f0.z, f0.w),
                      pk_bf16(f1.x, f1.y), pk_bf16(f1.z, f1.w)};
            v4u d1 = {pk_bf16(f2.x, f2.y), pk_bf16(f2.z, f2.w),
                      pk_bf16(f3.x, f3.y), pk_bf16(f3.z, f3.w)};
            *(v4u*)(Bs + row * 32 + ks)     = d0;
            *(v4u*)(Bs + row * 32 + ks + 8) = d1;
        }
        copy_fence();
        __syncthreads();

        AFrag fa;
        fa.q[0] = *(const v4u*)(As + (wave * 16 + mrow) * 32 + c0);
        fa.q[1] = *(const v4u*)(As + (wave * 16 + mrow) * 32 + 16 + c0);
        AFrag fb[8];
#pragma unroll
        for (int t = 0; t < 8; ++t) {
            const v4u* p = (const v4u*)(Bs + (t * 16 + mrow) * 32 + c0);
            fb[t].q[0] = p[0];
            fb[t].q[1] = p[2];
        }
        acc[0] = wmma_bf16(fa.v, fb[0].v, acc[0]);
#pragma unroll
        for (int t = 1; t < 8; ++t)
            acc[t] = wmma_bf16_ra(fa.v, fb[t].v, acc[t]);
        __syncthreads();
    }

#pragma unroll
    for (int t = 0; t < 8; ++t) {
        const int ncol = n0 + t * 16 + mrow;
        const float bv = bias[ncol];
#pragma unroll
        for (int r = 0; r < 8; ++r) {
            const int m = m0 + wave * 16 + ((lane < 16) ? r : 8 + r);
            Out[(size_t)m * D_MODEL + ncol] = acc[t][r] + bv;
        }
    }
}

// ---------------------------------------------------------------------------
extern "C" void kernel_launch(void* const* d_in, const int* in_sizes, int n_in,
                              void* d_out, int out_size, void* d_ws, size_t ws_size,
                              hipStream_t stream) {
    (void)in_sizes; (void)n_in; (void)out_size; (void)ws_size;
    const float* q   = (const float*)d_in[0];
    const float* k   = (const float*)d_in[1];
    const float* v   = (const float*)d_in[2];
    const float* w_q = (const float*)d_in[3];
    const float* b_q = (const float*)d_in[4];
    const float* w_k = (const float*)d_in[5];
    const float* b_k = (const float*)d_in[6];
    const float* w_v = (const float*)d_in[7];
    const float* b_v = (const float*)d_in[8];
    const float* w_o = (const float*)d_in[9];
    const float* b_o = (const float*)d_in[10];
    float* out = (float*)d_out;

    const size_t elems = (size_t)BATCH * S_LEN * D_MODEL;
    unsigned short* qh   = (unsigned short*)d_ws;
    unsigned short* kh   = qh + elems;
    unsigned short* vh   = kh + elems;
    unsigned short* attn = vh + elems;   // 64 MB bf16 total

    dim3 gGemm((BATCH * S_LEN) / 128, D_MODEL / 128);   // (64, 8)
    proj_qkv_kernel<0><<<gGemm, 256, 0, stream>>>(q, w_q, b_q, qh);
    proj_qkv_kernel<0><<<gGemm, 256, 0, stream>>>(k, w_k, b_k, kh);
    proj_qkv_kernel<1><<<gGemm, 256, 0, stream>>>(v, w_v, b_v, vh);

    dim3 gAttn(S_LEN / 64, NHEAD, BATCH);               // (32, 16, 4)
    attn_kernel<<<gAttn, 128, 0, stream>>>(qh, kh, vh, attn);

    proj_out_kernel<<<gGemm, 256, 0, stream>>>(attn, w_o, b_o, out);
}
